// RNNSent_3908420240126
// MI455X (gfx1250) — compile-verified, matrix operand order checked by reading, not run on hardware
//
#include <hip/hip_runtime.h>
#include <hip/hip_bf16.h>
#include <stdint.h>

// ---------------------------------------------------------------------------
// RNNSent (2-layer LSTM, T=300, B=1024, H=256) for gfx1250.
// GEMMs run as bf16 WMMA (v_wmma_f32_16x16x32_bf16) with f32 accumulation.
// ---------------------------------------------------------------------------

#define BATCH   1024
#define HID     256
#define TSTEPS  300
#define GCOLS   1024          // 4*HID gate columns

typedef __bf16 bf16x8  __attribute__((ext_vector_type(8)));
typedef __bf16 bf16x16 __attribute__((ext_vector_type(16)));
typedef float  f32x8   __attribute__((ext_vector_type(8)));

__device__ __forceinline__ unsigned short f2bf(float f) {
  unsigned int u = __float_as_uint(f);
  unsigned int r = u + 0x7FFFu + ((u >> 16) & 1u);   // round-to-nearest-even
  return (unsigned short)(r >> 16);
}

__device__ __forceinline__ f32x8 wmma_bf16f32(bf16x16 a, bf16x16 b, f32x8 c) {
#if defined(__gfx1250__)
  // (neg_a, A, neg_b, B, c_mod, C, reuse_a, reuse_b)
  return __builtin_amdgcn_wmma_f32_16x16x32_bf16(false, a, false, b, (short)0, c,
                                                 false, false);
#else
  (void)a; (void)b;
  return c;
#endif
}

__device__ __forceinline__ float sigmoidf_(float x) {
  return 1.0f / (1.0f + __expf(-x));
}

// ---------------------------------------------------------------------------
// Prep: bf16 weight panels. Wcat0: [1024 gates][288]  k<25 = W_ih0, 25..31 = 0,
// 32..287 = W_hh0.  Wcat1: [1024][512]  k<256 = W_ih1, else W_hh1.
// bsum = b_ih + b_hh (zeros in reference, kept for faithfulness).
// ---------------------------------------------------------------------------
__global__ void prep_weights_kernel(const float* __restrict__ Wih0,
                                    const float* __restrict__ Whh0,
                                    const float* __restrict__ bi0,
                                    const float* __restrict__ bh0,
                                    const float* __restrict__ Wih1,
                                    const float* __restrict__ Whh1,
                                    const float* __restrict__ bi1,
                                    const float* __restrict__ bh1,
                                    unsigned short* __restrict__ wcat0,
                                    unsigned short* __restrict__ wcat1,
                                    float* __restrict__ bs0,
                                    float* __restrict__ bs1) {
  const int N0 = GCOLS * 288, N1 = GCOLS * 512;
  const int total = N0 + N1 + GCOLS + GCOLS;
  for (int gid = blockIdx.x * blockDim.x + threadIdx.x; gid < total;
       gid += gridDim.x * blockDim.x) {
    if (gid < N0) {
      int row = gid / 288, k = gid % 288;
      float v = 0.0f;
      if (k < 25)       v = Wih0[row * 25 + k];
      else if (k >= 32) v = Whh0[row * 256 + (k - 32)];
      wcat0[gid] = f2bf(v);
    } else if (gid < N0 + N1) {
      int t = gid - N0;
      int row = t / 512, k = t % 512;
      float v = (k < 256) ? Wih1[row * 256 + k] : Whh1[row * 256 + (k - 256)];
      wcat1[t] = f2bf(v);
    } else if (gid < N0 + N1 + GCOLS) {
      int j = gid - (N0 + N1);
      bs0[j] = bi0[j] + bh0[j];
    } else {
      int j = gid - (N0 + N1 + GCOLS);
      bs1[j] = bi1[j] + bh1[j];
    }
  }
}

// ---------------------------------------------------------------------------
// Prep: gathered + reinterpreted embeddings, bf16, K padded 25->32.
// x[tt][b][k] = embflat[tt*25600 + b*25 + k],
// embflat[i]  = emb_table[input[bb][t]][e] with t=i/307200, bb=(i%307200)/300.
// ---------------------------------------------------------------------------
__global__ void prep_x_kernel(const int* __restrict__ input,
                              const float* __restrict__ emb,
                              unsigned short* __restrict__ xp) {
  const int total = TSTEPS * BATCH * 32;
  for (int gid = blockIdx.x * blockDim.x + threadIdx.x; gid < total;
       gid += gridDim.x * blockDim.x) {
    int tt  = gid >> 15;            // /(1024*32)
    int rem = gid & 32767;
    int b   = rem >> 5;
    int k   = rem & 31;
    float v = 0.0f;
    if (k < 25) {
      int flat = tt * 25600 + b * 25 + k;
      int t  = flat / 307200;
      int r2 = flat % 307200;
      int bb = r2 / 300;
      int e  = r2 % 300;
      int tok = input[bb * 25 + t];
      v = emb[(size_t)tok * 300 + e];
    }
    xp[gid] = f2bf(v);
  }
}

// ---------------------------------------------------------------------------
// Prep: initial h/c state (ping buffer 0).
// ---------------------------------------------------------------------------
__global__ void init_state_kernel(const float* __restrict__ h0,
                                  const float* __restrict__ c0,
                                  unsigned short* __restrict__ hbf0,
                                  unsigned short* __restrict__ hbf1,
                                  float* __restrict__ cb0,
                                  float* __restrict__ cb1,
                                  float* __restrict__ hf0,
                                  float* __restrict__ hf1) {
  const int per = BATCH * HID;      // 262144
  for (int gid = blockIdx.x * blockDim.x + threadIdx.x; gid < 2 * per;
       gid += gridDim.x * blockDim.x) {
    int l = gid >> 18;
    int idx = gid & (per - 1);
    float h = h0[gid], c = c0[gid];
    if (l == 0) { hbf0[idx] = f2bf(h); cb0[idx] = c; hf0[idx] = h; }
    else        { hbf1[idx] = f2bf(h); cb1[idx] = c; hf1[idx] = h; }
  }
}

// ---------------------------------------------------------------------------
// One LSTM layer timestep.
// Grid: (BATCH/32, HID/64), 256 threads = 8 waves.
// A rows (batch) come from src0 (first splitChunks K-chunks) then src1.
// B rows are Wcat gate rows; each WG covers gate cols {g*256+n0..+64, g=0..3}.
// LDS strides padded to 40 halfs / 260 f32 for bank-conflict-free access.
// ---------------------------------------------------------------------------
__global__ __launch_bounds__(256)
void lstm_step_kernel(const unsigned short* __restrict__ src0, int stride0,
                      int splitChunks,
                      const unsigned short* __restrict__ src1, int stride1,
                      const unsigned short* __restrict__ wcat, int strideW,
                      int nChunks,
                      const float* __restrict__ bsum,
                      float* __restrict__ cbuf,
                      float* __restrict__ hF32,
                      unsigned short* __restrict__ hBfOut) {
  __shared__ __align__(16) unsigned short ldsA[32 * 40];
  __shared__ __align__(16) unsigned short ldsB[256 * 40];
  __shared__ __align__(16) float ldsG[32 * 260];

  const int tid   = threadIdx.x;
  const int mbase = blockIdx.x * 32;   // batch-row block
  const int n0    = blockIdx.y * 64;   // hidden-unit block

  const int wv   = tid >> 5;
  const int g    = wv >> 1;            // gate index 0..3
  const int mt   = wv & 1;             // M tile 0..1
  const int lane = tid & 31;
  const int lm   = lane & 15;
  const int hi   = lane >> 4;

  f32x8 acc[4] = {};

  for (int kc = 0; kc < nChunks; ++kc) {
    // ---- stage A slab: 32 rows x 32 halfs (128 x 16B) ----
    if (tid < 128) {
      int r = tid >> 2, cj = tid & 3;
      int grow = mbase + r;
      const unsigned short* p;
      if (kc < splitChunks)
        p = src0 + (size_t)grow * stride0 + kc * 32 + cj * 8;
      else
        p = src1 + (size_t)grow * stride1 + (kc - splitChunks) * 32 + cj * 8;
      *(uint4*)&ldsA[r * 40 + cj * 8] = *(const uint4*)p;
    }
    // ---- stage B slab: 256 gate rows x 32 halfs (1024 x 16B) ----
#pragma unroll
    for (int it = 0; it < 4; ++it) {
      int q = tid + it * 256;
      int r = q >> 2, cj = q & 3;
      int gate = r >> 6, nl = r & 63;
      int wr = gate * 256 + n0 + nl;
      const unsigned short* p = wcat + (size_t)wr * strideW + kc * 32 + cj * 8;
      *(uint4*)&ldsB[r * 40 + cj * 8] = *(const uint4*)p;
    }
    __syncthreads();

    // ---- fragments + WMMA ----
    const unsigned short* pa = &ldsA[(mt * 16 + lm) * 40 + hi * 8];
    bf16x8 a0 = *(const bf16x8*)pa;
    bf16x8 a1 = *(const bf16x8*)(pa + 16);
    bf16x16 af = __builtin_shufflevector(a0, a1, 0, 1, 2, 3, 4, 5, 6, 7,
                                         8, 9, 10, 11, 12, 13, 14, 15);
#pragma unroll
    for (int nt = 0; nt < 4; ++nt) {
      const unsigned short* pb = &ldsB[(g * 64 + nt * 16 + lm) * 40 + hi * 8];
      bf16x8 b0 = *(const bf16x8*)pb;
      bf16x8 b1 = *(const bf16x8*)(pb + 16);
      bf16x16 bf = __builtin_shufflevector(b0, b1, 0, 1, 2, 3, 4, 5, 6, 7,
                                           8, 9, 10, 11, 12, 13, 14, 15);
      acc[nt] = wmma_bf16f32(af, bf, acc[nt]);
    }
    __syncthreads();
  }

  // ---- spill gate accumulators to LDS: (M=r+hi*8, N=lm) per ISA C layout ----
#pragma unroll
  for (int nt = 0; nt < 4; ++nt) {
    int col = g * 64 + nt * 16 + lm;
#pragma unroll
    for (int r = 0; r < 8; ++r) {
      int brow = mt * 16 + hi * 8 + r;
      ldsG[brow * 260 + col] = acc[nt][r];
    }
  }
  __syncthreads();

  // ---- fused LSTM cell: 2048 elements, 8 per thread ----
#pragma unroll
  for (int e = 0; e < 8; ++e) {
    int j  = e * 256 + tid;
    int bl = j >> 6;
    int nl = j & 63;
    int bg = mbase + bl;
    int ng = n0 + nl;
    float iv = ldsG[bl * 260 + nl]        + bsum[ng];
    float fv = ldsG[bl * 260 + 64 + nl]   + bsum[256 + ng];
    float gv = ldsG[bl * 260 + 128 + nl]  + bsum[512 + ng];
    float ov = ldsG[bl * 260 + 192 + nl]  + bsum[768 + ng];
    size_t idx = (size_t)bg * HID + ng;
    float cp = cbuf[idx];
    float cn = sigmoidf_(fv) * cp + sigmoidf_(iv) * tanhf(gv);
    float hn = sigmoidf_(ov) * tanhf(cn);
    cbuf[idx]   = cn;
    hF32[idx]   = hn;
    hBfOut[idx] = f2bf(hn);
  }
}

// ---------------------------------------------------------------------------
// Head: decoded = relu(h1_T @ fc_w.T + fc_b) @ dec_w.T + dec_b; store decoded.T
// ---------------------------------------------------------------------------
__global__ void head_kernel(const float* __restrict__ h1,
                            const float* __restrict__ fcw,
                            const float* __restrict__ fcb,
                            const float* __restrict__ decw,
                            const float* __restrict__ decb,
                            float* __restrict__ out) {
  int b = blockIdx.x * blockDim.x + threadIdx.x;
  if (b >= BATCH) return;
  float fc[10];
#pragma unroll
  for (int j = 0; j < 10; ++j) {
    float s = fcb[j];
    for (int k = 0; k < HID; ++k) s += h1[(size_t)b * HID + k] * fcw[j * HID + k];
    fc[j] = fmaxf(s, 0.0f);
  }
#pragma unroll
  for (int j = 0; j < 2; ++j) {
    float s = decb[j];
#pragma unroll
    for (int k = 0; k < 10; ++k) s += fc[k] * decw[j * 10 + k];
    out[j * BATCH + b] = s;   // decoded.T : [2, 1024]
  }
}

__global__ void copy_out_kernel(const float* __restrict__ hf0,
                                const float* __restrict__ hf1,
                                const float* __restrict__ cb0,
                                const float* __restrict__ cb1,
                                float* __restrict__ out) {
  const int per = BATCH * HID;  // 262144
  for (int gid = blockIdx.x * blockDim.x + threadIdx.x; gid < per;
       gid += gridDim.x * blockDim.x) {
    out[2048 + gid]                 = hf0[gid];
    out[2048 + per + gid]           = hf1[gid];
    out[2048 + 2 * per + gid]       = cb0[gid];
    out[2048 + 3 * per + gid]       = cb1[gid];
  }
}

// ---------------------------------------------------------------------------
extern "C" void kernel_launch(void* const* d_in, const int* in_sizes, int n_in,
                              void* d_out, int out_size, void* d_ws,
                              size_t ws_size, hipStream_t stream) {
  (void)in_sizes; (void)n_in; (void)out_size; (void)ws_size;

  const int*   input = (const int*)d_in[0];
  const float* h0    = (const float*)d_in[1];
  const float* c0    = (const float*)d_in[2];
  const float* emb   = (const float*)d_in[3];
  const float* fcw   = (const float*)d_in[4];
  const float* fcb   = (const float*)d_in[5];
  const float* decw  = (const float*)d_in[6];
  const float* decb  = (const float*)d_in[7];
  const float* Wih0  = (const float*)d_in[8];
  const float* Whh0  = (const float*)d_in[9];
  const float* bi0   = (const float*)d_in[10];
  const float* bh0   = (const float*)d_in[11];
  const float* Wih1  = (const float*)d_in[12];
  const float* Whh1  = (const float*)d_in[13];
  const float* bi1   = (const float*)d_in[14];
  const float* bh1   = (const float*)d_in[15];
  float* out = (float*)d_out;

  // ---- workspace carve-up ----
  char* base = (char*)d_ws;
  size_t off = 0;
  auto carve = [&](size_t bytes) {
    char* p = base + off;
    off = (off + bytes + 255) & ~(size_t)255;
    return p;
  };
  unsigned short* xp    = (unsigned short*)carve((size_t)TSTEPS * BATCH * 32 * 2);
  unsigned short* wcat0 = (unsigned short*)carve((size_t)GCOLS * 288 * 2);
  unsigned short* wcat1 = (unsigned short*)carve((size_t)GCOLS * 512 * 2);
  float* bs0 = (float*)carve(GCOLS * 4);
  float* bs1 = (float*)carve(GCOLS * 4);
  unsigned short* hbf0[2], *hbf1[2];
  hbf0[0] = (unsigned short*)carve((size_t)BATCH * HID * 2);
  hbf0[1] = (unsigned short*)carve((size_t)BATCH * HID * 2);
  hbf1[0] = (unsigned short*)carve((size_t)BATCH * HID * 2);
  hbf1[1] = (unsigned short*)carve((size_t)BATCH * HID * 2);
  float* hf0 = (float*)carve((size_t)BATCH * HID * 4);
  float* hf1 = (float*)carve((size_t)BATCH * HID * 4);
  float* cb0 = (float*)carve((size_t)BATCH * HID * 4);
  float* cb1 = (float*)carve((size_t)BATCH * HID * 4);

  // ---- one-time prep ----
  prep_weights_kernel<<<1024, 256, 0, stream>>>(Wih0, Whh0, bi0, bh0,
                                                Wih1, Whh1, bi1, bh1,
                                                wcat0, wcat1, bs0, bs1);
  {
    int n = TSTEPS * BATCH * 32;
    prep_x_kernel<<<(n + 255) / 256, 256, 0, stream>>>(input, emb, xp);
  }
  {
    int n = 2 * BATCH * HID;
    init_state_kernel<<<(n + 255) / 256, 256, 0, stream>>>(
        h0, c0, hbf0[0], hbf1[0], cb0, cb1, hf0, hf1);
  }

  // ---- 300 recurrent steps, 2 layers each ----
  dim3 grid(BATCH / 32, HID / 64);
  for (int t = 0; t < TSTEPS; ++t) {
    int pi = t & 1, po = 1 - pi;
    // layer 0: A = [x_t(32 padded) | h0_prev(256)], K = 288 (9 chunks)
    lstm_step_kernel<<<grid, 256, 0, stream>>>(
        xp + (size_t)t * BATCH * 32, 32, 1, hbf0[pi], HID,
        wcat0, 288, 9, bs0, cb0, hf0, hbf0[po]);
    // layer 1: A = [h0_t(256) | h1_prev(256)], K = 512 (16 chunks)
    lstm_step_kernel<<<grid, 256, 0, stream>>>(
        hbf0[po], HID, 8, hbf1[pi], HID,
        wcat1, 512, 16, bs1, cb1, hf1, hbf1[po]);
  }

  // ---- head + output assembly ----
  head_kernel<<<BATCH / 256, 256, 0, stream>>>(hf1, fcw, fcb, decw, decb, out);
  {
    int n = BATCH * HID;
    copy_out_kernel<<<(n + 255) / 256, 256, 0, stream>>>(hf0, hf1, cb0, cb1, out);
  }
}
